// TransformerBlock_14181982011878
// MI455X (gfx1250) — compile-verified, matrix operand order checked by reading, not run on hardware
//
#include <hip/hip_runtime.h>
#include <math.h>
#include <stdint.h>

#define H    896
#define NH   14
#define NKV  2
#define HD   64
#define DFF  4864
#define BATCH 2
#define SEQ  2048
#define T    (BATCH * SEQ)   /* 4096 tokens */
#define EPS  1e-5f
#define SCALE 0.125f         /* 1/sqrt(64) */

typedef __bf16 bf16;
typedef __attribute__((ext_vector_type(16))) __bf16 v16bf;
typedef __attribute__((ext_vector_type(8)))  __bf16 bf16x8;
typedef __attribute__((ext_vector_type(8)))  float   v8f;

union FragB { v16bf v; bf16x8 h[2]; };

__device__ inline v8f wmma_bf16(v16bf a, v16bf b, v8f c) {
  return __builtin_amdgcn_wmma_f32_16x16x32_bf16(
      /*neg_a=*/false, a, /*neg_b=*/false, b,
      /*c_mod=*/(short)0, c, /*reuse_a=*/false, /*reuse_b=*/false);
}

/* ---------------- elementwise f32 -> bf16 cast ---------------- */
__global__ void cvt_f32_bf16(const float* __restrict__ src, bf16* __restrict__ dst, int n) {
  int i = blockIdx.x * blockDim.x + threadIdx.x;
  if (i < n) dst[i] = (bf16)src[i];
}

/* ---------------- RMSNorm (fp32 in, bf16 out), one block per token ---------------- */
__global__ void rmsnorm_bf16(const float* __restrict__ x, const float* __restrict__ w,
                             bf16* __restrict__ out) {
  __shared__ float red[128];
  int t = blockIdx.x;
  const float* row = x + (size_t)t * H;
  float ss = 0.f;
  for (int i = threadIdx.x; i < H; i += 128) { float v = row[i]; ss += v * v; }
  red[threadIdx.x] = ss;
  __syncthreads();
  for (int s = 64; s > 0; s >>= 1) {
    if ((int)threadIdx.x < s) red[threadIdx.x] += red[threadIdx.x + s];
    __syncthreads();
  }
  float inv = rsqrtf(red[0] / (float)H + EPS);
  bf16* orow = out + (size_t)t * H;
  for (int i = threadIdx.x; i < H; i += 128)
    orow[i] = (bf16)(row[i] * inv * w[i]);
}

/* ---------------- generic WMMA GEMM: C[M,N] = A[M,K] * W[N,K]^T (+bias)
   MODE 0: f32 store        MODE 1: bf16 store (+bias)
   MODE 2: f32 store + residual   MODE 3: bf16 transposed store (+bias)  (for V^T)
   block = 64 (2 waves). Block tile 32(M) x 128(N); wave tile 32 x 64.
   A tile (32x32 bf16, double-buffered) staged in LDS via GLOBAL_LOAD_ASYNC_TO_LDS,
   overlapped with WMMA compute; all 4 B fragments loaded before the 8-WMMA burst
   so a single loadcnt wait covers the group. grid = (M/32, N/128). */
template <int MODE>
__global__ void __launch_bounds__(64, 1)
gemm_bf16(const bf16* __restrict__ A, const bf16* __restrict__ W,
          const float* __restrict__ bias, const float* __restrict__ resid,
          float* __restrict__ Cf, bf16* __restrict__ Cb,
          int M, int N, int K) {
  __shared__ __align__(16) bf16 tileA[2][32][32];

  int tid  = threadIdx.x;          // 0..63
  int lane = tid & 31;
  int wave = tid >> 5;             // 0..1
  int m0 = blockIdx.x * 32;
  int n0 = blockIdx.y * 128 + wave * 64;
  int lm = lane & 15;
  int klo  = (lane < 16) ? 0 : 8;  // 16-bit A/B fragment K pattern per ISA layout
  int base = (lane < 16) ? 0 : 8;

  const bf16* wrow = W + (size_t)(n0 + lm) * K;

  v8f acc[2][4];
#pragma unroll
  for (int mi = 0; mi < 2; mi++)
#pragma unroll
    for (int j = 0; j < 4; j++) acc[mi][j] = v8f{};

  /* async-stage one 32x32 A tile (2048B): 64 threads x 2 b128 transfers */
  auto issue_async = [&](int buf, int k) {
#pragma unroll
    for (int i = 0; i < 2; i++) {
      int e   = tid + 64 * i;       // 0..127
      int row = e >> 2;             // 0..31
      int seg = e & 3;              // 0..3 (8 bf16 = 16B each)
      unsigned lds = (unsigned)(uintptr_t)(&tileA[buf][row][seg * 8]);
      unsigned long long ga =
          (unsigned long long)(uintptr_t)(A + (size_t)(m0 + row) * K + k + seg * 8);
      asm volatile("global_load_async_to_lds_b128 %0, %1, off"
                   :: "v"(lds), "v"(ga) : "memory");
    }
  };

  issue_async(0, 0);
  for (int k = 0; k < K; k += 32) {
    int buf = (k >> 5) & 1;
    if (k + 32 < K) {
      issue_async(buf ^ 1, k + 32);                      // prefetch next tile
      asm volatile("s_wait_asynccnt 0x2" ::: "memory");  // current tile landed
    } else {
      asm volatile("s_wait_asynccnt 0x0" ::: "memory");
    }
    __syncthreads();

    __builtin_prefetch(wrow + k + 512, 0, 1);            // global_prefetch_b8

    /* load ALL fragments first (2 ds_load_b128 x2 + 8 global_load_b128), then
       run the 8-WMMA burst: one wait covers the group, WMMAs issue back-to-back */
    FragB a[2];
#pragma unroll
    for (int mi = 0; mi < 2; mi++) {
      const bf16* arow = &tileA[buf][mi * 16 + lm][0];
      a[mi].h[0] = *(const bf16x8*)(arow + klo);          // ds_load_b128
      a[mi].h[1] = *(const bf16x8*)(arow + klo + 16);
    }
    FragB b[4];
#pragma unroll
    for (int j = 0; j < 4; j++) {
      const bf16* wj = wrow + (size_t)(16 * j) * K + k;
      b[j].h[0] = *(const bf16x8*)(wj + klo);
      b[j].h[1] = *(const bf16x8*)(wj + klo + 16);
    }
#pragma unroll
    for (int j = 0; j < 4; j++) {
      acc[0][j] = wmma_bf16(a[0].v, b[j].v, acc[0][j]);   // B frag reused x2
      acc[1][j] = wmma_bf16(a[1].v, b[j].v, acc[1][j]);
    }
    __syncthreads();   // all waves done reading before tile buffer is refilled
  }

#pragma unroll
  for (int mi = 0; mi < 2; mi++) {
#pragma unroll
    for (int j = 0; j < 4; j++) {
      int c = n0 + 16 * j + lm;
      float bi = bias ? bias[c] : 0.f;
#pragma unroll
      for (int r = 0; r < 8; r++) {
        int m = m0 + mi * 16 + r + base;   // C layout: row = r + 8*(lane>=16)
        float v = acc[mi][j][r] + bi;
        if (MODE == 0)      Cf[(size_t)m * N + c] = v;
        else if (MODE == 1) Cb[(size_t)m * N + c] = (bf16)v;
        else if (MODE == 2) Cf[(size_t)m * N + c] = v + resid[(size_t)m * N + c];
        else                Cb[(size_t)c * M + m] = (bf16)v;   // transposed (V^T)
      }
    }
  }
}

/* ---------------- flash attention: one wave per (b, head, 16-query tile).
   __launch_bounds__(32,1): single wave owns the SIMD -> full VGPR budget,
   keeps Q/K/V fragments + O accumulators resident (no scratch spills). */
__global__ void __launch_bounds__(32, 1)
attn_kernel(const bf16* __restrict__ Q,   // [T, NH*HD]
            const bf16* __restrict__ Kc,  // [T, NKV*HD]
            const bf16* __restrict__ VT,  // [NKV*HD, T]
            const float* __restrict__ mask, // [B,1,S,S]
            bf16* __restrict__ O) {       // [T, NH*HD]
  int lane = threadIdx.x;
  int qt = blockIdx.x;
  int hh = blockIdx.y;
  int b  = blockIdx.z;
  int hkv = hh / (NH / NKV);
  int q0 = b * SEQ + qt * 16;
  int lm = lane & 15;
  int klo  = (lane < 16) ? 0 : 8;
  int base = (lane < 16) ? 0 : 8;

  __shared__ float sc[16][32];
  __shared__ float s_alpha[16];
  __shared__ float s_l[16];

  /* Q A-fragments for the two 32-wide d-chunks of HD=64 */
  const bf16* qrow = Q + (size_t)(q0 + lm) * H + hh * HD;
  FragB qa0, qa1;
  qa0.h[0] = *(const bf16x8*)(qrow + klo);
  qa0.h[1] = *(const bf16x8*)(qrow + klo + 16);
  qa1.h[0] = *(const bf16x8*)(qrow + 32 + klo);
  qa1.h[1] = *(const bf16x8*)(qrow + 32 + klo + 16);

  v8f o[4] = {v8f{}, v8f{}, v8f{}, v8f{}};
  float m_i = -1e30f, l_i = 0.f;            // valid in lanes 0..15 (row = lane)
  const float* mbase = mask + (size_t)b * SEQ * SEQ;

  for (int kt = 0; kt < SEQ; kt += 32) {
    /* ---- scores: load all 4 K fragments, then 4 WMMAs back-to-back ---- */
    FragB kb[2][2];
#pragma unroll
    for (int g = 0; g < 2; g++) {
      int keytok = b * SEQ + kt + g * 16 + lm;
      const bf16* krow = Kc + (size_t)keytok * (NKV * HD) + hkv * HD;
      kb[g][0].h[0] = *(const bf16x8*)(krow + klo);
      kb[g][0].h[1] = *(const bf16x8*)(krow + klo + 16);
      kb[g][1].h[0] = *(const bf16x8*)(krow + 32 + klo);
      kb[g][1].h[1] = *(const bf16x8*)(krow + 32 + klo + 16);
    }
    v8f s[2];
#pragma unroll
    for (int g = 0; g < 2; g++) {
      s[g] = wmma_bf16(qa0.v, kb[g][0].v, v8f{});
      s[g] = wmma_bf16(qa1.v, kb[g][1].v, s[g]);
    }

    /* ---- hoist V-fragment loads: independent of softmax, latency hidden ---- */
    FragB vb[4];
#pragma unroll
    for (int c = 0; c < 4; c++) {
      const bf16* vcol = VT + (size_t)(hkv * HD + c * 16 + lm) * T + b * SEQ + kt;
      vb[c].h[0] = *(const bf16x8*)(vcol + klo);
      vb[c].h[1] = *(const bf16x8*)(vcol + klo + 16);
    }

    /* scale + mask + spill scores to LDS */
#pragma unroll
    for (int g = 0; g < 2; g++) {
      int col = kt + g * 16 + lm;
#pragma unroll
      for (int r = 0; r < 8; r++) {
        int qr = qt * 16 + r + base;
        sc[r + base][g * 16 + lm] = s[g][r] * SCALE + mbase[(size_t)qr * SEQ + col];
      }
    }
    __syncthreads();

    /* online softmax stats, lanes 0..15 own one query row each */
    if (lane < 16) {
      float mx = m_i;
#pragma unroll
      for (int j = 0; j < 32; j++) mx = fmaxf(mx, sc[lane][j]);
      float alpha = __expf(m_i - mx);
      float sum = 0.f;
#pragma unroll
      for (int j = 0; j < 32; j++) {
        float p = __expf(sc[lane][j] - mx);
        sc[lane][j] = p;
        sum += p;
      }
      l_i = l_i * alpha + sum;
      m_i = mx;
      s_alpha[lane] = alpha;
    }
    __syncthreads();

    /* repack P (16x32) into A-fragment layout, fp32 -> bf16 */
    FragB pa;
    {
      const float* prow = &sc[lm][0];
      bf16 tmp[16];
#pragma unroll
      for (int e = 0; e < 8; e++) tmp[e] = (bf16)prow[klo + e];
#pragma unroll
      for (int e = 0; e < 8; e++) tmp[8 + e] = (bf16)prow[klo + 16 + e];
      pa.h[0] = *(bf16x8*)&tmp[0];
      pa.h[1] = *(bf16x8*)&tmp[8];
    }

    /* rescale O accumulators, then PV: 4 WMMAs back-to-back */
#pragma unroll
    for (int c = 0; c < 4; c++) {
#pragma unroll
      for (int r = 0; r < 8; r++) o[c][r] *= s_alpha[r + base];
    }
#pragma unroll
    for (int c = 0; c < 4; c++) o[c] = wmma_bf16(pa.v, vb[c].v, o[c]);
    __syncthreads();   // protect sc before next iteration's writes
  }

  if (lane < 16) s_l[lane] = l_i;
  __syncthreads();
#pragma unroll
  for (int c = 0; c < 4; c++) {
#pragma unroll
    for (int r = 0; r < 8; r++) {
      float inv = 1.f / s_l[r + base];
      O[(size_t)(q0 + r + base) * H + hh * HD + c * 16 + lm] = (bf16)(o[c][r] * inv);
    }
  }
}

/* ---------------- SwiGLU combine: gu = silu(gate) * up ---------------- */
__global__ void silu_mul(const bf16* __restrict__ g, const bf16* __restrict__ u,
                         bf16* __restrict__ out, int n) {
  int i = blockIdx.x * blockDim.x + threadIdx.x;
  if (i < n) {
    float gv = (float)g[i];
    float s = gv / (1.f + __expf(-gv));
    out[i] = (bf16)(s * (float)u[i]);
  }
}

/* ---------------- orchestration ---------------- */
extern "C" void kernel_launch(void* const* d_in, const int* in_sizes, int n_in,
                              void* d_out, int out_size, void* d_ws, size_t ws_size,
                              hipStream_t stream) {
  const float* x    = (const float*)d_in[0];
  /* d_in[1] = position_ids (unused by reference) */
  const float* mask = (const float*)d_in[2];
  const float* ln1  = (const float*)d_in[3];
  const float* qw   = (const float*)d_in[4];
  const float* qb   = (const float*)d_in[5];
  const float* kw   = (const float*)d_in[6];
  const float* kb   = (const float*)d_in[7];
  const float* vw   = (const float*)d_in[8];
  const float* vb   = (const float*)d_in[9];
  const float* ow   = (const float*)d_in[10];
  const float* ln2  = (const float*)d_in[11];
  const float* gw   = (const float*)d_in[12];
  const float* uw   = (const float*)d_in[13];
  const float* dw   = (const float*)d_in[14];
  float* out = (float*)d_out;

  char* ws = (char*)d_ws;
  size_t off = 0;
  auto alloc = [&](size_t bytes) -> char* {
    char* p = ws + off;
    off += (bytes + 255) & ~(size_t)255;
    return p;
  };

  bf16* h1    = (bf16*)alloc((size_t)T * H * 2);
  bf16* h2    = (bf16*)alloc((size_t)T * H * 2);
  bf16* qact  = (bf16*)alloc((size_t)T * H * 2);          // NH*HD == H
  bf16* kact  = (bf16*)alloc((size_t)T * (NKV * HD) * 2);
  bf16* vTact = (bf16*)alloc((size_t)(NKV * HD) * T * 2); // transposed V
  bf16* attn  = (bf16*)alloc((size_t)T * H * 2);
  float* x1   = (float*)alloc((size_t)T * H * 4);
  bf16* gact  = (bf16*)alloc((size_t)T * DFF * 2);
  bf16* uact  = (bf16*)alloc((size_t)T * DFF * 2);
  bf16* guact = (bf16*)alloc((size_t)T * DFF * 2);
  bf16* qwb = (bf16*)alloc((size_t)H * H * 2);
  bf16* kwb = (bf16*)alloc((size_t)(NKV * HD) * H * 2);
  bf16* vwb = (bf16*)alloc((size_t)(NKV * HD) * H * 2);
  bf16* owb = (bf16*)alloc((size_t)H * H * 2);
  bf16* gwb = (bf16*)alloc((size_t)DFF * H * 2);
  bf16* uwb = (bf16*)alloc((size_t)DFF * H * 2);
  bf16* dwb = (bf16*)alloc((size_t)H * DFF * 2);
  (void)ws_size; (void)in_sizes; (void)n_in; (void)out_size;

  auto cvt = [&](const float* s, bf16* d, int n) {
    cvt_f32_bf16<<<(n + 255) / 256, 256, 0, stream>>>(s, d, n);
  };
  cvt(qw, qwb, H * H);
  cvt(kw, kwb, NKV * HD * H);
  cvt(vw, vwb, NKV * HD * H);
  cvt(ow, owb, H * H);
  cvt(gw, gwb, DFF * H);
  cvt(uw, uwb, DFF * H);
  cvt(dw, dwb, H * DFF);

  /* pre-norm attention */
  rmsnorm_bf16<<<T, 128, 0, stream>>>(x, ln1, h1);

  gemm_bf16<1><<<dim3(T / 32, H / 128), 64, 0, stream>>>(
      h1, qwb, qb, nullptr, nullptr, qact, T, H, H);
  gemm_bf16<1><<<dim3(T / 32, (NKV * HD) / 128), 64, 0, stream>>>(
      h1, kwb, kb, nullptr, nullptr, kact, T, NKV * HD, H);
  gemm_bf16<3><<<dim3(T / 32, (NKV * HD) / 128), 64, 0, stream>>>(
      h1, vwb, vb, nullptr, nullptr, vTact, T, NKV * HD, H);

  attn_kernel<<<dim3(SEQ / 16, NH, BATCH), 32, 0, stream>>>(
      qact, kact, vTact, mask, attn);

  /* o-proj + residual -> x1 */
  gemm_bf16<2><<<dim3(T / 32, H / 128), 64, 0, stream>>>(
      attn, owb, nullptr, x, x1, nullptr, T, H, H);

  /* pre-norm SwiGLU MLP */
  rmsnorm_bf16<<<T, 128, 0, stream>>>(x1, ln2, h2);

  gemm_bf16<1><<<dim3(T / 32, DFF / 128), 64, 0, stream>>>(
      h2, gwb, nullptr, nullptr, nullptr, gact, T, DFF, H);
  gemm_bf16<1><<<dim3(T / 32, DFF / 128), 64, 0, stream>>>(
      h2, uwb, nullptr, nullptr, nullptr, uact, T, DFF, H);

  silu_mul<<<((size_t)T * DFF + 255) / 256, 256, 0, stream>>>(
      gact, uact, guact, T * DFF);

  /* down-proj + residual -> out */
  gemm_bf16<2><<<dim3(T / 32, H / 128), 64, 0, stream>>>(
      guact, dwb, nullptr, x1, out, nullptr, T, H, DFF);
}
